// Attention_9199819948120
// MI455X (gfx1250) — matrix operand order repaired and checked by reading, not compiled, not run on hardware
//
#include <hip/hip_runtime.h>
#include <hip/hip_bf16.h>

// ---------------------------------------------------------------------------
// Attention block for MI455X (gfx1250, wave32, WMMA).
//   q/k/v = X @ Wqkv + b  -> rotary -> flash attention (causal) -> @ W_O + b_O
// All matmuls use v_wmma_f32_16x16x32_bf16 (bf16 inputs, f32 accumulation).
// ---------------------------------------------------------------------------

typedef unsigned short u16;
typedef __attribute__((ext_vector_type(16))) __bf16 bf16x16;
typedef __attribute__((ext_vector_type(8)))  float  f32x8;
typedef int v4i_g __attribute__((vector_size(16)));   // matches builtin's V4i

#define BATCH   2
#define SEQ     2048
#define DMODEL  2048
#define NHEADS  16
#define DHEAD   128
#define ROWS    (BATCH * SEQ)          // 4096
#define HE      (NHEADS * DHEAD)       // 2048

union AFrag { bf16x16 v; uint4 u[2]; u16 s[16]; };

__device__ __forceinline__ u16 f2bf(float x) {
  unsigned u = __float_as_uint(x);
  unsigned r = u + 0x7FFFu + ((u >> 16) & 1u);   // round-to-nearest-even
  return (u16)(r >> 16);
}

// ---- async global->LDS copy (CDNA5) with safe fallback --------------------
#if defined(__gfx1250__) && __has_builtin(__builtin_amdgcn_global_load_async_to_lds_b128)
#define HAVE_ASYNC_LDS 1
#endif

__device__ __forceinline__ void cp_b128(u16* ldst, const u16* gsrc) {
#if HAVE_ASYNC_LDS
  __builtin_amdgcn_global_load_async_to_lds_b128(
      (__attribute__((address_space(1))) v4i_g*)gsrc,
      (__attribute__((address_space(3))) v4i_g*)ldst, 0, 0);
#else
  *(uint4*)ldst = *(const uint4*)gsrc;
#endif
}

__device__ __forceinline__ void cp_wait() {
#if HAVE_ASYNC_LDS
#if __has_builtin(__builtin_amdgcn_s_wait_asynccnt)
  __builtin_amdgcn_s_wait_asynccnt(0);
#else
  asm volatile("s_wait_asynccnt 0" ::: "memory");
#endif
#endif
}

// ---------------------------------------------------------------------------
// Elementwise conversion kernels
// ---------------------------------------------------------------------------
__global__ void cvt_f32_to_bf16(const float* __restrict__ x, u16* __restrict__ y, int n) {
  int i = blockIdx.x * blockDim.x + threadIdx.x;
  if (i < n) y[i] = f2bf(x[i]);
}

// W[h][d][e] -> Wt[d][h*DHEAD+e]   (so projections are plain row-major GEMMs)
__global__ void cvt_wqkv_bf16(const float* __restrict__ w, u16* __restrict__ y) {
  int i = blockIdx.x * blockDim.x + threadIdx.x;   // over NHEADS*DMODEL*DHEAD
  const int e = i & (DHEAD - 1);
  const int d = (i >> 7) & (DMODEL - 1);
  const int h = i >> 18;
  y[(size_t)d * HE + h * DHEAD + e] = f2bf(w[i]);
}

// Rotary (rotate-half, ROTARY_DIM == DHEAD == 128) on q,k (f32), convert all to bf16.
__global__ void rotary_and_cvt(const float* __restrict__ qf, const float* __restrict__ kf,
                               const float* __restrict__ vf,
                               u16* __restrict__ qb, u16* __restrict__ kb, u16* __restrict__ vb) {
  const int i = blockIdx.x * blockDim.x + threadIdx.x;  // over ROWS*HE
  const int e   = i & (DHEAD - 1);
  const int pos = (i >> 11) & (SEQ - 1);                // i = ((b*S+s)*H + h)*E + e
  const int j   = e & 63;
  const float freq = __powf(10000.0f, (float)j * (1.0f / 64.0f));
  const float ang  = (float)pos / freq;
  float sn, cs;
  __sincosf(ang, &sn, &cs);
  const int ip = (i & ~(DHEAD - 1)) | ((e + 64) & (DHEAD - 1));
  const float xq = qf[i], xqp = qf[ip];
  const float xk = kf[i], xkp = kf[ip];
  const float rq = (e < 64) ? -xqp : xqp;
  const float rk = (e < 64) ? -xkp : xkp;
  qb[i] = f2bf(xq * cs + rq * sn);
  kb[i] = f2bf(xk * cs + rk * sn);
  vb[i] = f2bf(vf[i]);
}

// ---------------------------------------------------------------------------
// Tiled WMMA GEMM: C[M,N] = A[M,K](bf16,row) * B[K,N](bf16,row) + bias[N]
// Block: 128x128, 8 waves (2x4), each wave: 4x2 tiles of 16x16, K-step 32.
// LDS: A staged [m][k] (async), B staged transposed [n][k].
// ---------------------------------------------------------------------------
__global__ __launch_bounds__(256)
void gemm_bf16_wmma(const u16* __restrict__ A, const u16* __restrict__ Bm,
                    float* __restrict__ C, const float* __restrict__ bias,
                    int M, int N, int K) {
  __shared__ u16 As[128 * 32];
  __shared__ u16 Bs[128 * 32];

  const int tid  = threadIdx.x;
  const int wave = tid >> 5, lane = tid & 31;
  const int wrow = wave >> 2, wcol = wave & 3;
  const int m0 = blockIdx.y * 128, n0 = blockIdx.x * 128;
  const int lm = lane & 15;
  const int hi = lane >> 4;        // half-wave select per WMMA layout
  const int lk8 = hi * 8, lk16 = hi * 16;

  f32x8 acc[4][2];
  const f32x8 z8 = {0.f, 0.f, 0.f, 0.f, 0.f, 0.f, 0.f, 0.f};
#pragma unroll
  for (int i = 0; i < 4; ++i)
#pragma unroll
    for (int j = 0; j < 2; ++j) acc[i][j] = z8;

  const int ar = tid >> 1, ac = (tid & 1) * 16;   // A stage: row, col-seg
  const int bk = tid >> 3, bn = (tid & 7) * 16;   // B stage: k-row, n-seg

  for (int k0 = 0; k0 < K; k0 += 32) {
    // stage A tile (async -> LDS)
    const u16* gA = A + (size_t)(m0 + ar) * K + k0 + ac;
    cp_b128(&As[ar * 32 + ac], gA);
    cp_b128(&As[ar * 32 + ac + 8], gA + 8);
    __builtin_prefetch(gA + 32, 0, 1);
    // stage B tile transposed into [n][k]
    const u16* gB = Bm + (size_t)(k0 + bk) * N + n0 + bn;
    uint4 b0 = *(const uint4*)gB;
    uint4 b1 = *(const uint4*)(gB + 8);
    __builtin_prefetch(gB + (size_t)32 * N, 0, 1);
    const u16* p0 = (const u16*)&b0;
    const u16* p1 = (const u16*)&b1;
#pragma unroll
    for (int t = 0; t < 8; ++t) {
      Bs[(bn + t) * 32 + bk]     = p0[t];
      Bs[(bn + 8 + t) * 32 + bk] = p1[t];
    }
    cp_wait();
    __syncthreads();

    AFrag a[4], bf[2];
#pragma unroll
    for (int mi = 0; mi < 4; ++mi) {
      const int r = wrow * 64 + mi * 16 + lm;
      a[mi].u[0] = *(const uint4*)&As[r * 32 + lk8];        // K = lk8..+7
      a[mi].u[1] = *(const uint4*)&As[r * 32 + 16 + lk8];   // K = 16+lk8..+7
    }
#pragma unroll
    for (int ni = 0; ni < 2; ++ni) {
      const int c = wcol * 32 + ni * 16 + lm;
      bf[ni].u[0] = *(const uint4*)&Bs[c * 32 + lk8];       // K = lk8..+7
      bf[ni].u[1] = *(const uint4*)&Bs[c * 32 + 16 + lk8];  // K = 16+lk8..+7
    }
#pragma unroll
    for (int mi = 0; mi < 4; ++mi)
#pragma unroll
      for (int ni = 0; ni < 2; ++ni)
        acc[mi][ni] = __builtin_amdgcn_wmma_f32_16x16x32_bf16(
            false, a[mi].v, false, bf[ni].v, (short)0, acc[mi][ni], false, false);
    __syncthreads();
  }

  // epilogue: C-layout -> global f32 + bias
#pragma unroll
  for (int mi = 0; mi < 4; ++mi) {
#pragma unroll
    for (int ni = 0; ni < 2; ++ni) {
      const int cm = m0 + wrow * 64 + mi * 16 + hi * 8;
      const int cn = n0 + wcol * 32 + ni * 16 + lm;
      const float bv = bias[cn];
#pragma unroll
      for (int j = 0; j < 8; ++j)
        C[(size_t)(cm + j) * N + cn] = acc[mi][ni][j] + bv;
    }
  }
}

// ---------------------------------------------------------------------------
// Flash attention (causal). One workgroup = one (b,h) x 128-row Q block.
// 8 waves * 16 query rows. LDS (dynamic 96KB): K[key][e], V^T[e][key], Q/P.
// ---------------------------------------------------------------------------
__global__ __launch_bounds__(256)
void flash_attn_wmma(const u16* __restrict__ Q, const u16* __restrict__ Kt,
                     const u16* __restrict__ V, u16* __restrict__ Z) {
  extern __shared__ u16 smem[];
  u16* Ks = smem;                 // [128 keys][128 e]
  u16* Vs = smem + 128 * 128;     // [128 e][128 keys] (transposed)
  u16* Ps = smem + 2 * 128 * 128; // Q tile first, then P tile

  const int tid = threadIdx.x, wave = tid >> 5, lane = tid & 31;
  const int lm = lane & 15, hi = lane >> 4;
  const int lk8 = hi * 8, lk16 = hi * 16;
  const int qb = blockIdx.x, bh = blockIdx.y;
  const int b = bh >> 4, h = bh & 15;
  const int q0 = qb * 128;
  const float inv_scale = 0.08838834764831845f;   // 1/sqrt(128)

  // stage Q block [row][e]
  {
    const int r = tid >> 1;
    const int e0 = (tid & 1) * 64;
    const u16* gq = Q + ((size_t)(b * SEQ + q0 + r) * NHEADS + h) * DHEAD + e0;
#pragma unroll
    for (int u = 0; u < 8; ++u) cp_b128(&Ps[r * 128 + e0 + u * 8], gq + u * 8);
  }
  cp_wait();
  __syncthreads();

  // Q A-fragments (16 rows per wave, E split into 4 k-steps of 32)
  AFrag aQ[4];
#pragma unroll
  for (int kk = 0; kk < 4; ++kk) {
    const int r = wave * 16 + lm;
    aQ[kk].u[0] = *(const uint4*)&Ps[r * 128 + kk * 32 + lk8];
    aQ[kk].u[1] = *(const uint4*)&Ps[r * 128 + kk * 32 + 16 + lk8];
  }
  __syncthreads();   // Ps now reusable for P

  const f32x8 z8 = {0.f, 0.f, 0.f, 0.f, 0.f, 0.f, 0.f, 0.f};
  f32x8 o[8];
#pragma unroll
  for (int i = 0; i < 8; ++i) o[i] = z8;
  float rm[8], rl[8];
#pragma unroll
  for (int j = 0; j < 8; ++j) { rm[j] = -3.0e38f; rl[j] = 0.0f; }

  for (int kb = 0; kb <= qb; ++kb) {
    // stage K [key][e] (async) and V transposed [e][key]
    {
      const int r = tid >> 1;
      const int e0 = (tid & 1) * 64;
      const size_t base = ((size_t)(b * SEQ + kb * 128 + r) * NHEADS + h) * DHEAD + e0;
      const u16* gk = Kt + base;
#pragma unroll
      for (int u = 0; u < 8; ++u) cp_b128(&Ks[r * 128 + e0 + u * 8], gk + u * 8);
      const u16* gv = V + base;
#pragma unroll
      for (int u = 0; u < 8; ++u) {
        uint4 vv = *(const uint4*)(gv + u * 8);
        const u16* vs = (const u16*)&vv;
#pragma unroll
        for (int t = 0; t < 8; ++t) Vs[(e0 + u * 8 + t) * 128 + r] = vs[t];
      }
    }
    cp_wait();
    __syncthreads();

    // S = Q @ K^T  (8 key-subtiles of 16, k over E)
    f32x8 st[8];
#pragma unroll
    for (int nt = 0; nt < 8; ++nt) {
      f32x8 s = z8;
#pragma unroll
      for (int kk = 0; kk < 4; ++kk) {
        AFrag bK;
        const int n = nt * 16 + lm;   // key column of B
        bK.u[0] = *(const uint4*)&Ks[n * 128 + kk * 32 + lk8];
        bK.u[1] = *(const uint4*)&Ks[n * 128 + kk * 32 + 16 + lk8];
        s = __builtin_amdgcn_wmma_f32_16x16x32_bf16(
            false, aQ[kk].v, false, bK.v, (short)0, s, false, false);
      }
      st[nt] = s;
    }

    // scale + causal mask (only needed on the diagonal block)
    const bool diag = (kb == qb);
#pragma unroll
    for (int nt = 0; nt < 8; ++nt)
#pragma unroll
      for (int j = 0; j < 8; ++j) {
        float sv = st[nt][j] * inv_scale;
        if (diag) {
          const int key = kb * 128 + nt * 16 + lm;
          const int qr  = q0 + wave * 16 + hi * 8 + j;
          if (key > qr) sv = -3.0e38f;
        }
        st[nt][j] = sv;
      }

    // online softmax (row lives across 16 lanes of one half-wave)
#pragma unroll
    for (int j = 0; j < 8; ++j) {
      float bm = -3.0e38f;
#pragma unroll
      for (int nt = 0; nt < 8; ++nt) bm = fmaxf(bm, st[nt][j]);
#pragma unroll
      for (int off = 1; off < 16; off <<= 1) bm = fmaxf(bm, __shfl_xor(bm, off, 32));
      const float mnew = fmaxf(rm[j], bm);
      const float corr = __expf(rm[j] - mnew);
      rm[j] = mnew;
      float rs = 0.0f;
#pragma unroll
      for (int nt = 0; nt < 8; ++nt) {
        const float p = __expf(st[nt][j] - mnew);
        st[nt][j] = p;
        rs += p;
      }
#pragma unroll
      for (int off = 1; off < 16; off <<= 1) rs += __shfl_xor(rs, off, 32);
      rl[j] = rl[j] * corr + rs;
#pragma unroll
      for (int et = 0; et < 8; ++et) o[et][j] *= corr;
    }

    // P -> LDS as bf16 (per-wave private rows)
#pragma unroll
    for (int nt = 0; nt < 8; ++nt)
#pragma unroll
      for (int j = 0; j < 8; ++j) {
        const int r = wave * 16 + hi * 8 + j;
        Ps[r * 128 + nt * 16 + lm] = f2bf(st[nt][j]);
      }
    __syncthreads();

    // O += P @ V
#pragma unroll
    for (int et = 0; et < 8; ++et)
#pragma unroll
      for (int kk = 0; kk < 4; ++kk) {
        AFrag aP, bV;
        const int r = wave * 16 + lm;
        aP.u[0] = *(const uint4*)&Ps[r * 128 + kk * 32 + lk8];
        aP.u[1] = *(const uint4*)&Ps[r * 128 + kk * 32 + 16 + lk8];
        const int e = et * 16 + lm;
        bV.u[0] = *(const uint4*)&Vs[e * 128 + kk * 32 + lk8];
        bV.u[1] = *(const uint4*)&Vs[e * 128 + kk * 32 + 16 + lk8];
        o[et] = __builtin_amdgcn_wmma_f32_16x16x32_bf16(
            false, aP.v, false, bV.v, (short)0, o[et], false, false);
      }
    __syncthreads();
  }

  // normalize + store Z (bf16, layout [b,s,h,e])
#pragma unroll
  for (int j = 0; j < 8; ++j) {
    const float inv = 1.0f / rl[j];
    const int qr = q0 + wave * 16 + hi * 8 + j;
    const size_t base = ((size_t)(b * SEQ + qr) * NHEADS + h) * DHEAD;
#pragma unroll
    for (int et = 0; et < 8; ++et)
      Z[base + et * 16 + lm] = f2bf(o[et][j] * inv);
  }
}

// ---------------------------------------------------------------------------
// Host launcher
// ---------------------------------------------------------------------------
extern "C" void kernel_launch(void* const* d_in, const int* in_sizes, int n_in,
                              void* d_out, int out_size, void* d_ws, size_t ws_size,
                              hipStream_t stream) {
  (void)in_sizes; (void)n_in; (void)out_size; (void)ws_size;
  const float* xq = (const float*)d_in[0];
  const float* xk = (const float*)d_in[1];
  const float* xv = (const float*)d_in[2];
  const float* WQ = (const float*)d_in[3];
  const float* WK = (const float*)d_in[4];
  const float* WV = (const float*)d_in[5];
  const float* WO = (const float*)d_in[6];
  const float* bQ = (const float*)d_in[7];
  const float* bK = (const float*)d_in[8];
  const float* bV = (const float*)d_in[9];
  const float* bO = (const float*)d_in[10];
  float* out = (float*)d_out;

  const size_t NX = (size_t)ROWS * DMODEL;    // 8,388,608 activations
  const size_t NW = (size_t)DMODEL * DMODEL;  // 4,194,304 weights

  char* ws = (char*)d_ws;
  u16* xq_bf = (u16*)ws; ws += NX * 2;
  u16* xk_bf = (u16*)ws; ws += NX * 2;
  u16* xv_bf = (u16*)ws; ws += NX * 2;
  u16* wq_bf = (u16*)ws; ws += NW * 2;
  u16* wk_bf = (u16*)ws; ws += NW * 2;
  u16* wv_bf = (u16*)ws; ws += NW * 2;
  u16* wo_bf = (u16*)ws; ws += NW * 2;
  float* qf  = (float*)ws; ws += NX * 4;
  float* kf  = (float*)ws; ws += NX * 4;
  float* vf  = (float*)ws; ws += NX * 4;
  u16* qbh   = (u16*)ws; ws += NX * 2;
  u16* kbh   = (u16*)ws; ws += NX * 2;
  u16* vbh   = (u16*)ws; ws += NX * 2;
  u16* zbh   = (u16*)qf;               // alias: qf is dead after rotary

  const int T = 256;
  // 1) conversions
  cvt_f32_to_bf16<<<(int)(NX / T), T, 0, stream>>>(xq, xq_bf, (int)NX);
  cvt_f32_to_bf16<<<(int)(NX / T), T, 0, stream>>>(xk, xk_bf, (int)NX);
  cvt_f32_to_bf16<<<(int)(NX / T), T, 0, stream>>>(xv, xv_bf, (int)NX);
  cvt_wqkv_bf16 <<<(int)(NW / T), T, 0, stream>>>(WQ, wq_bf);
  cvt_wqkv_bf16 <<<(int)(NW / T), T, 0, stream>>>(WK, wk_bf);
  cvt_wqkv_bf16 <<<(int)(NW / T), T, 0, stream>>>(WV, wv_bf);
  cvt_f32_to_bf16<<<(int)(NW / T), T, 0, stream>>>(WO, wo_bf, (int)NW);

  // 2) QKV projections  (M=4096, N=2048, K=2048)
  dim3 gg(HE / 128, ROWS / 128);
  gemm_bf16_wmma<<<gg, T, 0, stream>>>(xq_bf, wq_bf, qf, bQ, ROWS, HE, DMODEL);
  gemm_bf16_wmma<<<gg, T, 0, stream>>>(xk_bf, wk_bf, kf, bK, ROWS, HE, DMODEL);
  gemm_bf16_wmma<<<gg, T, 0, stream>>>(xv_bf, wv_bf, vf, bV, ROWS, HE, DMODEL);

  // 3) rotary + bf16
  rotary_and_cvt<<<(int)(NX / T), T, 0, stream>>>(qf, kf, vf, qbh, kbh, vbh);

  // 4) flash attention: grid = (S/128 q-blocks, B*H)
  dim3 gf(SEQ / 128, BATCH * NHEADS);
  flash_attn_wmma<<<gf, T, 3 * 128 * 128 * 2, stream>>>(qbh, kbh, vbh, zbh);

  // 5) output projection: out = Z @ W_O + b_O
  dim3 go(DMODEL / 128, ROWS / 128);
  gemm_bf16_wmma<<<go, T, 0, stream>>>(zbh, wo_bf, out, bO, ROWS, DMODEL, HE);
}
